// GATv2Model_25366076850812
// MI455X (gfx1250) — matrix operand order repaired and checked by reading, not compiled, not run on hardware
//
#include <hip/hip_runtime.h>

#define NNODES 50000
#define NEDGES 1600000
#define NGR 8

typedef __bf16 bf16_t;
typedef __attribute__((ext_vector_type(16))) __bf16 v16bf;
typedef __attribute__((ext_vector_type(8)))  __bf16 v8bf;
typedef __attribute__((ext_vector_type(8)))  float  v8f;

__device__ __forceinline__ float lrelu(float x, float s) { return x > 0.f ? x : s * x; }

// ---------------------------------------------------------------------------
// WMMA GEMM:  C[M x 64] = A[M x 64] (bf16) @ B[64 x 64] (bf16)
// B pre-swizzled into fragment-major layout (8 tiles x 32 lanes x 16
// contiguous bf16): each fragment is one aligned 32B vector load.
// Output: f32 (+bias) if Cb==nullptr, else bf16 to Cb.
// 256 threads = 8 waves; each wave computes a 16x64 strip (8 WMMAs).
// All indices 32-bit (buffers < 2^31 bytes).
// ---------------------------------------------------------------------------
__global__ __launch_bounds__(256) void k_gemm64(const bf16_t* __restrict__ A,
                                                const v16bf* __restrict__ Bsw,
                                                const float* __restrict__ bias,
                                                float* __restrict__ Cf,
                                                bf16_t* __restrict__ Cb, unsigned M)
{
    const int      wave    = threadIdx.x >> 5;
    const int      lane    = threadIdx.x & 31;
    const unsigned rowbase = blockIdx.x * 128u + wave * 16u;

    const int mr = lane & 15;            // M row (A) / N col (B,C) within tile
    const int kh = lane >> 4;            // K-half (A) / K-group (B) / M-half (C)

    unsigned arow = rowbase + mr;
    if (arow >= M) arow = M - 1u;        // clamp loads; stores predicated below

    // A fragments in contiguous K order (k = 16*kh + s), matching Bsw: two 16B chunks per fragment.
    const v8bf* ap8 = (const v8bf*)(A + arow * 64u);
    v8bf l0 = ap8[2 * kh],     h0 = ap8[2 * kh + 1];
    v8bf l1 = ap8[4 + 2 * kh], h1 = ap8[5 + 2 * kh];
    v16bf a0 = __builtin_shufflevector(l0, h0, 0,1,2,3,4,5,6,7,8,9,10,11,12,13,14,15);
    v16bf a1 = __builtin_shufflevector(l1, h1, 0,1,2,3,4,5,6,7,8,9,10,11,12,13,14,15);

    // Preload all 8 B fragments (tile t = kt*4+nt), one 32B vector each.
    v16bf bfrag[8];
#pragma unroll
    for (int t = 0; t < 8; ++t) bfrag[t] = Bsw[t * 32 + lane];

    v8f acc[4];
#pragma unroll
    for (int nt = 0; nt < 4; ++nt) { v8f z = {}; acc[nt] = z; }

#pragma unroll
    for (int kt = 0; kt < 2; ++kt) {
        v16bf a = kt ? a1 : a0;
#pragma unroll
        for (int nt = 0; nt < 4; ++nt)
            acc[nt] = __builtin_amdgcn_wmma_f32_16x16x32_bf16(
                false, a, false, bfrag[kt * 4 + nt], (short)0, acc[nt], false, false);
    }

    // C layout: lanes 0-15 -> rows v+0..7, lanes 16-31 -> rows v+8..15; N = mr.
    // Uniform fast path: tile fully in-range (always true for M % 128 == 0).
    const bool     full = (rowbase + 16u) <= M;
    const unsigned rb   = rowbase + (unsigned)(kh * 8);
    if (Cb) {
        if (full) {
#pragma unroll
            for (int nt = 0; nt < 4; ++nt) {
                const int col = nt * 16 + mr;
#pragma unroll
                for (int v = 0; v < 8; ++v)
                    Cb[(rb + v) * 64u + col] = (bf16_t)acc[nt][v];
            }
        } else {
#pragma unroll
            for (int nt = 0; nt < 4; ++nt) {
                const int col = nt * 16 + mr;
#pragma unroll
                for (int v = 0; v < 8; ++v)
                    if (rb + v < M) Cb[(rb + v) * 64u + col] = (bf16_t)acc[nt][v];
            }
        }
    } else {
        if (full) {
#pragma unroll
            for (int nt = 0; nt < 4; ++nt) {
                const int   col = nt * 16 + mr;
                const float bv  = bias[col];
#pragma unroll
                for (int v = 0; v < 8; ++v)
                    Cf[(rb + v) * 64u + col] = acc[nt][v] + bv;
            }
        } else {
#pragma unroll
            for (int nt = 0; nt < 4; ++nt) {
                const int   col = nt * 16 + mr;
                const float bv  = bias[col];
#pragma unroll
                for (int v = 0; v < 8; ++v)
                    if (rb + v < M) Cf[(rb + v) * 64u + col] = acc[nt][v] + bv;
            }
        }
    }
}

// ---------------------------------------------------------------------------
// Convert 64x64 f32 weights -> bf16 in WMMA fragment-major (swizzled) layout:
// dst[((kt*4+nt)*32 + lane)*16 + j] = B[kt*32 + (lane>>4)*16 + j][nt*16 + (lane&15)]
// ---------------------------------------------------------------------------
__global__ void k_convert_wsw(const float* __restrict__ src, bf16_t* __restrict__ dst)
{
    unsigned D = blockIdx.x * 256u + threadIdx.x;
    if (D >= 4096u) return;
    unsigned j  = D & 15u;
    unsigned l  = (D >> 4) & 31u;
    unsigned t  = D >> 9;
    unsigned kt = t >> 2, nt = t & 3u;
    unsigned K  = kt * 32u + (l >> 4) * 16u + j;
    unsigned n  = nt * 16u + (l & 15u);
    dst[D] = (bf16_t)src[K * 64u + n];
}

// ---------------------------------------------------------------------------
// xh = x @ node_W + node_b  (N x 16 @ 16 x 64); also emit bf16 copy
// ---------------------------------------------------------------------------
__global__ __launch_bounds__(256) void k_node_emb(const float* __restrict__ x,
                                                  const float* __restrict__ W,
                                                  const float* __restrict__ b,
                                                  float* __restrict__ xh,
                                                  bf16_t* __restrict__ xhb, unsigned n)
{
    unsigned t = blockIdx.x * 256u + threadIdx.x;
    if (t >= n * 64u) return;
    unsigned row = t >> 6; unsigned ch = t & 63u;
    float acc = b[ch];
#pragma unroll
    for (int k = 0; k < 16; ++k) acc += x[row * 16u + k] * W[k * 64u + ch];
    xh[t]  = acc;
    xhb[t] = (bf16_t)acc;
}

// ---------------------------------------------------------------------------
// ea = leaky(edge_attr @ eemb_W + b, 0.01) -> bf16; accumulate column sums
// (spread across 256 partial slots to avoid hot atomics)
// ---------------------------------------------------------------------------
__global__ __launch_bounds__(256) void k_edge_emb(const float* __restrict__ eattr,
                                                  const float* __restrict__ W,
                                                  const float* __restrict__ b,
                                                  bf16_t* __restrict__ eab,
                                                  float* __restrict__ partial, unsigned E)
{
    __shared__ float red[256];
    unsigned t = blockIdx.x * 256u + threadIdx.x;
    unsigned edge = t >> 6; unsigned ch = t & 63u;
    float v = 0.f;
    if (edge < E) {
        float acc = b[ch];
#pragma unroll
        for (int k = 0; k < 8; ++k) acc += eattr[edge * 8u + k] * W[k * 64u + ch];
        acc = lrelu(acc, 0.01f);
        eab[edge * 64u + ch] = (bf16_t)acc;
        v = acc;
    }
    red[threadIdx.x] = v;
    __syncthreads();
    if (threadIdx.x < 64) {
        float s = red[threadIdx.x] + red[threadIdx.x + 64] +
                  red[threadIdx.x + 128] + red[threadIdx.x + 192];
        atomicAdd(&partial[(blockIdx.x & 255u) * 64u + threadIdx.x], s);
    }
}

__global__ void k_mean(const float* __restrict__ partial, float* __restrict__ mean, float invE)
{
    int ch = threadIdx.x;   // 64 threads
    float s = 0.f;
    for (int q = 0; q < 256; ++q) s += partial[q * 64 + ch];
    mean[ch] = s * invE;
}

// eself[64] = mean_ea[64] @ Wed[64x64]  (self-loop edge transform), bf16 out
__global__ void k_vecmat64(const float* __restrict__ vec, const float* __restrict__ W,
                           bf16_t* __restrict__ out)
{
    int ch = threadIdx.x;   // 64 threads
    float acc = 0.f;
    for (int k = 0; k < 64; ++k) acc += vec[k] * W[k * 64 + ch];
    out[ch] = (bf16_t)acc;
}

__global__ void k_init_out(const float* __restrict__ cb, float* __restrict__ outb, unsigned n)
{
    unsigned t = blockIdx.x * 256u + threadIdx.x;
    if (t >= n * 64u) return;
    outb[t] = cb[t & 63u];
}

// ---------------------------------------------------------------------------
// Pass 1: alpha = <leaky(xl[src]+xr[dst]+e, 0.2), att> per head; atomic max/dst
// float-max via monotone uint key: neg -> ~u, pos -> u|0x80000000
// ---------------------------------------------------------------------------
__global__ __launch_bounds__(256) void k_edge_alpha(
    const float* __restrict__ xl, const float* __restrict__ xr,
    const bf16_t* __restrict__ ebuf, const bf16_t* __restrict__ eself,
    const int* __restrict__ srcs, const int* __restrict__ dsts,
    const float* __restrict__ att, float* __restrict__ alphab,
    unsigned* __restrict__ amaxK, unsigned E, unsigned n, int shift)
{
    unsigned i = blockIdx.x * 256u + threadIdx.x;
    unsigned Etot = E + n;
    if (i >= Etot) return;
    unsigned s, d; const bf16_t* er;
    if (i < E) { s = (unsigned)srcs[i]; d = (unsigned)dsts[i]; er = ebuf + i * 64u; }
    else       { s = d = i - E;                                er = eself; }
    const float4* pl = (const float4*)(xl + s * 64u);
    const float4* pr = (const float4*)(xr + d * 64u);
    const v8bf*   pe = (const v8bf*)er;
    const float4* pw = (const float4*)att;
    float acc[4] = {0.f, 0.f, 0.f, 0.f};
#pragma unroll
    for (int q = 0; q < 8; ++q) {
        v8bf ev = pe[q];
#pragma unroll
        for (int half = 0; half < 2; ++half) {
            float4 a  = pl[q * 2 + half];
            float4 b  = pr[q * 2 + half];
            float4 wv = pw[q * 2 + half];
            int ch = q * 8 + half * 4;
            float m;
            m = a.x + b.x + (float)ev[half * 4 + 0]; m = m > 0.f ? m : 0.2f * m;
            acc[(ch + 0) >> shift] += m * wv.x;
            m = a.y + b.y + (float)ev[half * 4 + 1]; m = m > 0.f ? m : 0.2f * m;
            acc[(ch + 1) >> shift] += m * wv.y;
            m = a.z + b.z + (float)ev[half * 4 + 2]; m = m > 0.f ? m : 0.2f * m;
            acc[(ch + 2) >> shift] += m * wv.z;
            m = a.w + b.w + (float)ev[half * 4 + 3]; m = m > 0.f ? m : 0.2f * m;
            acc[(ch + 3) >> shift] += m * wv.w;
        }
    }
    const int h = 64 >> shift;
#pragma unroll
    for (int hd = 0; hd < 4; ++hd) {
        if (hd < h) {
            float v = acc[hd];
            alphab[i * 4u + hd] = v;
            unsigned u   = __float_as_uint(v);
            unsigned key = (u & 0x80000000u) ? ~u : (u | 0x80000000u);
            atomicMax(&amaxK[d * 4u + hd], key);
        }
    }
}

// Pass 2: ex = exp(alpha - amax[dst]); den[dst] += ex (in-place overwrite alpha)
__global__ __launch_bounds__(256) void k_edge_den(
    float* __restrict__ alphab, const unsigned* __restrict__ amaxK,
    float* __restrict__ den, const int* __restrict__ dsts,
    unsigned E, unsigned n, int h)
{
    unsigned i = blockIdx.x * 256u + threadIdx.x;
    if (i >= E + n) return;
    unsigned d = (i < E) ? (unsigned)dsts[i] : (i - E);
    for (int hd = 0; hd < h; ++hd) {
        unsigned key = amaxK[d * 4u + hd];
        unsigned u   = (key & 0x80000000u) ? (key & 0x7FFFFFFFu) : ~key;
        float amax = __uint_as_float(u);
        float ex = __expf(alphab[i * 4u + hd] - amax);
        alphab[i * 4u + hd] = ex;
        atomicAdd(&den[d * 4u + hd], ex);
    }
}

// Pass 3: out[dst,ch] += (ex/den[dst]) * xl[src,ch]; one thread per (edge,ch)
__global__ __launch_bounds__(256) void k_edge_aggr(
    const float* __restrict__ exb, const float* __restrict__ den,
    const float* __restrict__ xl, const int* __restrict__ srcs,
    const int* __restrict__ dsts, float* __restrict__ outb,
    unsigned E, unsigned n, int shift)
{
    unsigned t = blockIdx.x * 256u + threadIdx.x;
    unsigned tot = (E + n) * 64u;
    if (t >= tot) return;
    unsigned i = t >> 6; unsigned ch = t & 63u;
    unsigned s, d;
    if (i < E) { s = (unsigned)srcs[i]; d = (unsigned)dsts[i]; }
    else       { s = d = i - E; }
    unsigned hd = ch >> shift;
    float a = exb[i * 4u + hd] / den[d * 4u + hd];
    atomicAdd(&outb[d * 64u + ch], a * xl[s * 64u + ch]);
}

// ---------------------------------------------------------------------------
// BatchNorm (batch statistics) + leaky, emitting f32 + bf16 activations
// ---------------------------------------------------------------------------
__global__ __launch_bounds__(256) void k_bn_stats(const float* __restrict__ xin,
                                                  float* __restrict__ sums, unsigned n)
{
    __shared__ float rs[256], rq[256];
    const unsigned ch  = threadIdx.x & 63u;
    const unsigned sub = threadIdx.x >> 6;
    unsigned start = blockIdx.x * 256u;
    unsigned end   = start + 256u; if (end > n) end = n;
    float s = 0.f, q = 0.f;
    for (unsigned r = start + sub; r < end; r += 4u) {
        float v = xin[r * 64u + ch];
        s += v; q += v * v;
    }
    rs[threadIdx.x] = s; rq[threadIdx.x] = q;
    __syncthreads();
    if (threadIdx.x < 128) {
        rs[threadIdx.x] += rs[threadIdx.x + 128];
        rq[threadIdx.x] += rq[threadIdx.x + 128];
    }
    __syncthreads();
    if (threadIdx.x < 64) {
        atomicAdd(&sums[threadIdx.x],      rs[threadIdx.x] + rs[threadIdx.x + 64]);
        atomicAdd(&sums[64 + threadIdx.x], rq[threadIdx.x] + rq[threadIdx.x + 64]);
    }
}

__global__ __launch_bounds__(256) void k_bn_apply(const float* __restrict__ xin,
                                                  const float* __restrict__ sums,
                                                  const float* __restrict__ g,
                                                  const float* __restrict__ bt,
                                                  float* __restrict__ xh,
                                                  bf16_t* __restrict__ xhb, unsigned n)
{
    unsigned t = blockIdx.x * 256u + threadIdx.x;
    if (t >= n * 64u) return;
    unsigned ch = t & 63u;
    float invn = 1.f / (float)n;
    float mu  = sums[ch] * invn;
    float var = sums[64 + ch] * invn - mu * mu;
    float y = (xin[t] - mu) * rsqrtf(var + 1e-5f) * g[ch] + bt[ch];
    y = lrelu(y, 0.01f);
    xh[t]  = y;
    xhb[t] = (bf16_t)y;
}

__global__ __launch_bounds__(256) void k_pool(const float* __restrict__ xh,
                                              const int* __restrict__ batch,
                                              float* __restrict__ pooled, unsigned n)
{
    unsigned t = blockIdx.x * 256u + threadIdx.x;
    if (t >= n * 64u) return;
    unsigned row = t >> 6; unsigned ch = t & 63u;
    atomicAdd(&pooled[(unsigned)batch[row] * 64u + ch], xh[t]);
}

__global__ __launch_bounds__(256) void k_readout(const float* __restrict__ pooled,
                                                 const float* __restrict__ W1,
                                                 const float* __restrict__ b1,
                                                 const float* __restrict__ W2,
                                                 const float* __restrict__ b2,
                                                 float* __restrict__ out)
{
    __shared__ float hid[NGR * 32];
    int g = threadIdx.x >> 5, j = threadIdx.x & 31;   // 8 graphs x 32 hidden
    float acc = b1[j];
    for (int k = 0; k < 64; ++k) acc += pooled[g * 64 + k] * W1[k * 32 + j];
    hid[g * 32 + j] = lrelu(acc, 0.01f);
    __syncthreads();
    if (threadIdx.x < NGR) {
        float o = b2[0];
        for (int jj = 0; jj < 32; ++jj) o += hid[threadIdx.x * 32 + jj] * W2[jj];
        out[threadIdx.x] = o;
    }
}

// ---------------------------------------------------------------------------
extern "C" void kernel_launch(void* const* d_in, const int* in_sizes, int n_in,
                              void* d_out, int out_size, void* d_ws, size_t ws_size,
                              hipStream_t stream)
{
    (void)in_sizes; (void)n_in; (void)out_size; (void)ws_size;

    const float* x      = (const float*)d_in[0];
    const int*   eidx   = (const int*)d_in[1];
    const float* eattr  = (const float*)d_in[2];
    const int*   batch  = (const int*)d_in[3];
    const float* node_W = (const float*)d_in[4];
    const float* node_b = (const float*)d_in[5];
    const float* eemb_W = (const float*)d_in[6];
    const float* eemb_b = (const float*)d_in[7];
    const float *Wl[3], *bl[3], *Wr[3], *br[3], *Wed[3], *att[3], *cb[3], *gg[3], *bt[3];
    for (int i = 0; i < 3; ++i) {
        const int base = 8 + i * 9;
        Wl[i]  = (const float*)d_in[base + 0]; bl[i] = (const float*)d_in[base + 1];
        Wr[i]  = (const float*)d_in[base + 2]; br[i] = (const float*)d_in[base + 3];
        Wed[i] = (const float*)d_in[base + 4]; att[i] = (const float*)d_in[base + 5];
        cb[i]  = (const float*)d_in[base + 6]; gg[i] = (const float*)d_in[base + 7];
        bt[i]  = (const float*)d_in[base + 8];
    }
    const float* ro1_W = (const float*)d_in[35];
    const float* ro1_b = (const float*)d_in[36];
    const float* ro2_W = (const float*)d_in[37];
    const float* ro2_b = (const float*)d_in[38];

    const unsigned N = NNODES, E = NEDGES;
    const unsigned Etot = E + N;
    const int* srcs = eidx;
    const int* dsts = eidx + E;

    char* w = (char*)d_ws;
    auto alloc = [&](size_t bytes) -> char* {
        char* p = w; w += (bytes + 255) & ~(size_t)255; return p;
    };
    bf16_t*   ea_bf  = (bf16_t*)  alloc((size_t)E * 64 * sizeof(bf16_t));
    bf16_t*   ebuf   = (bf16_t*)  alloc((size_t)E * 64 * sizeof(bf16_t));
    float*    xh     = (float*)   alloc((size_t)N * 64 * sizeof(float));
    bf16_t*   xh_bf  = (bf16_t*)  alloc((size_t)N * 64 * sizeof(bf16_t));
    float*    xl     = (float*)   alloc((size_t)N * 64 * sizeof(float));
    float*    xr     = (float*)   alloc((size_t)N * 64 * sizeof(float));
    float*    outb   = (float*)   alloc((size_t)N * 64 * sizeof(float));
    float*    alphab = (float*)   alloc((size_t)Etot * 4 * sizeof(float));
    unsigned* amaxK  = (unsigned*)alloc((size_t)N * 4 * sizeof(unsigned));
    float*    den    = (float*)   alloc((size_t)N * 4 * sizeof(float));
    float*    partial= (float*)   alloc(256 * 64 * sizeof(float));
    float*    meanv  = (float*)   alloc(64 * sizeof(float));
    bf16_t*   eself  = (bf16_t*)  alloc(64 * sizeof(bf16_t));
    float*    bnsum  = (float*)   alloc(128 * sizeof(float));
    float*    pooled = (float*)   alloc(NGR * 64 * sizeof(float));
    bf16_t*   Wl_bf  = (bf16_t*)  alloc(4096 * sizeof(bf16_t));
    bf16_t*   Wr_bf  = (bf16_t*)  alloc(4096 * sizeof(bf16_t));
    bf16_t*   Wed_bf = (bf16_t*)  alloc(4096 * sizeof(bf16_t));

    const unsigned nth = N * 64u;
    const unsigned eth = E * 64u;

    // edge embedding (+ column mean), node embedding
    hipMemsetAsync(partial, 0, 256 * 64 * sizeof(float), stream);
    k_edge_emb<<<(eth + 255u) / 256u, 256, 0, stream>>>(eattr, eemb_W, eemb_b,
                                                        ea_bf, partial, E);
    k_mean<<<1, 64, 0, stream>>>(partial, meanv, 1.0f / (float)E);
    k_node_emb<<<(nth + 255u) / 256u, 256, 0, stream>>>(x, node_W, node_b, xh, xh_bf, N);

    const int shifts[3] = {4, 4, 6};   // c = 16,16,64
    const int heads[3]  = {4, 4, 1};
    for (int L = 0; L < 3; ++L) {
        k_convert_wsw<<<16, 256, 0, stream>>>(Wl[L],  Wl_bf);
        k_convert_wsw<<<16, 256, 0, stream>>>(Wr[L],  Wr_bf);
        k_convert_wsw<<<16, 256, 0, stream>>>(Wed[L], Wed_bf);
        k_vecmat64<<<1, 64, 0, stream>>>(meanv, Wed[L], eself);

        k_gemm64<<<(N + 127u) / 128u, 256, 0, stream>>>(xh_bf, (const v16bf*)Wl_bf,
                                                        bl[L], xl, nullptr, N);
        k_gemm64<<<(N + 127u) / 128u, 256, 0, stream>>>(xh_bf, (const v16bf*)Wr_bf,
                                                        br[L], xr, nullptr, N);
        k_gemm64<<<(E + 127u) / 128u, 256, 0, stream>>>(ea_bf, (const v16bf*)Wed_bf,
                                                        nullptr, nullptr, ebuf, E);

        hipMemsetAsync(amaxK, 0, (size_t)N * 4 * sizeof(unsigned), stream);
        hipMemsetAsync(den,   0, (size_t)N * 4 * sizeof(float), stream);
        k_init_out<<<(nth + 255u) / 256u, 256, 0, stream>>>(cb[L], outb, N);

        k_edge_alpha<<<(Etot + 255u) / 256u, 256, 0, stream>>>(
            xl, xr, ebuf, eself, srcs, dsts, att[L], alphab, amaxK, E, N, shifts[L]);
        k_edge_den<<<(Etot + 255u) / 256u, 256, 0, stream>>>(
            alphab, amaxK, den, dsts, E, N, heads[L]);
        const unsigned ath = Etot * 64u;
        k_edge_aggr<<<(ath + 255u) / 256u, 256, 0, stream>>>(
            alphab, den, xl, srcs, dsts, outb, E, N, shifts[L]);

        hipMemsetAsync(bnsum, 0, 128 * sizeof(float), stream);
        k_bn_stats<<<(N + 255u) / 256u, 256, 0, stream>>>(outb, bnsum, N);
        k_bn_apply<<<(nth + 255u) / 256u, 256, 0, stream>>>(outb, bnsum, gg[L], bt[L],
                                                            xh, xh_bf, N);
    }

    hipMemsetAsync(pooled, 0, NGR * 64 * sizeof(float), stream);
    k_pool<<<(nth + 255u) / 256u, 256, 0, stream>>>(xh, batch, pooled, N);
    k_readout<<<1, 256, 0, stream>>>(pooled, ro1_W, ro1_b, ro2_W, ro2_b, (float*)d_out);
}
